// BDCLoss_74594991997728
// MI455X (gfx1250) — compile-verified, matrix operand order checked by reading, not run on hardware
//
#include <hip/hip_runtime.h>

// ---------------------------------------------------------------------------
// BDC loss for MI455X (gfx1250, wave32, WMMA)
//   intra : per-row reduction (bandwidth trivial vs 23.3 TB/s)
//   adv   : fhat @ fhat^T hinge, 8192x8192x1024 == 137 GFLOP,
//           fp16 operands (16 MB, L2-resident) -> v_wmma_f32_16x16x32_f16.
//   GEMM tiling: 128x128 block / 4 wave32 waves, 64x64 per wave (4x4 WMMA
//   tiles, 1:1 ds_load_b128 : wmma), BK=64, double-buffered LDS with
//   async global->LDS copies (ASYNCcnt) when the builtin is available.
//   Upper-triangular blocks only (symmetric matrix, weight 2 off-diagonal).
// ---------------------------------------------------------------------------

typedef __attribute__((ext_vector_type(16))) _Float16 v16h;
typedef __attribute__((ext_vector_type(8)))  float    v8f;
typedef __attribute__((ext_vector_type(4)))  int      v4i;

#define B_    8192
#define D_    1024
#define C_    1000
#define BM    128
#define BK    64
#define LDSK  72            // padded LDS row stride in halfs (144 B, 16B-aligned frags)
#define NTILE (B_ / BM)     // 64
#define NBLK  (NTILE * NTILE)

#if __has_builtin(__builtin_amdgcn_global_load_async_to_lds_b128)
#define ASYNC_OK 1
typedef __attribute__((address_space(1))) v4i* glb_v4i_p;
typedef __attribute__((address_space(3))) v4i* lds_v4i_p;
#endif

__device__ __forceinline__ float wave_sum(float v) {
#pragma unroll
  for (int o = 16; o > 0; o >>= 1) v += __shfl_down(v, o, 32);
  return v;
}
__device__ __forceinline__ double wave_sum_d(double v) {
#pragma unroll
  for (int o = 16; o > 0; o >>= 1) v += __shfl_down(v, o, 32);
  return v;
}

// 16-bit A/B fragment load per CDNA5 ISA 7.12.2:
// lane L: row = L&15, kbase = (L>>4)*8; halfs[0..7]=K kbase..kbase+7,
// halfs[8..15]=K kbase+16..kbase+23.
__device__ __forceinline__ v16h load_frag(const _Float16* tile, int lane) {
  const int m  = lane & 15;
  const int kb = (lane >> 4) * 8;
  const _Float16* p = tile + m * LDSK + kb;
  union { v16h h; uint4 u[2]; } f;
  f.u[0] = *reinterpret_cast<const uint4*>(p);
  f.u[1] = *reinterpret_cast<const uint4*>(p + 16);
  return f.h;
}

// Fill one 128x64-half LDS tile from global. 128 threads, 8 chunks (16 B)
// per thread; lane-consecutive chunk ids -> 4 rows x 128 contiguous bytes
// per 32-lane instruction (coalesced).
__device__ __forceinline__ void stage_load(const _Float16* __restrict__ gbase,
                                           _Float16* lbase, int t) {
#pragma unroll
  for (int q = 0; q < 8; ++q) {
    const int id  = t + 128 * q;
    const int row = id >> 3;
    const int col = (id & 7) * 8;
    const _Float16* g = gbase + (size_t)row * D_ + col;
    _Float16*       l = lbase + row * LDSK + col;
#ifdef ASYNC_OK
    __builtin_amdgcn_global_load_async_to_lds_b128((glb_v4i_p)g, (lds_v4i_p)l, 0, 0);
#else
    *reinterpret_cast<uint4*>(l) = *reinterpret_cast<const uint4*>(g);
#endif
  }
}

__global__ void bdc_zero_hist(int* __restrict__ hist) {
  hist[threadIdx.x] = 0;   // launched with 1024 threads, hist padded to 1024
}

// ---------------------------------------------------------------------------
// Per-row kernel: norms, cosine sim, squared error, fp16 normalized features.
// ---------------------------------------------------------------------------
__global__ __launch_bounds__(256) void bdc_row(
    const float* __restrict__ feat, const int* __restrict__ labels,
    const float* __restrict__ centers, _Float16* __restrict__ fhat,
    float* __restrict__ intra_partial, int* __restrict__ hist) {
  const int i = blockIdx.x;
  const int t = threadIdx.x;
  const int lab = labels[i];
  const float* f = feat + (size_t)i * D_;
  const float* c = centers + (size_t)lab * D_;

  float ff = 0.f, cc = 0.f, fc = 0.f, se = 0.f;
#pragma unroll
  for (int d = t; d < D_; d += 256) {
    const float fv = f[d], cv = c[d];
    ff = fmaf(fv, fv, ff);
    cc = fmaf(cv, cv, cc);
    fc = fmaf(fv, cv, fc);
    const float e = fv - cv;
    se = fmaf(e, e, se);
  }
  ff = wave_sum(ff); cc = wave_sum(cc); fc = wave_sum(fc); se = wave_sum(se);

  __shared__ float4 sred[8];
  __shared__ float  s_inv;
  const int wave = t >> 5, lane = t & 31;
  if (lane == 0) sred[wave] = make_float4(ff, cc, fc, se);
  __syncthreads();
  if (t == 0) {
    float4 s = sred[0];
#pragma unroll
    for (int w = 1; w < 8; ++w) {
      s.x += sred[w].x; s.y += sred[w].y; s.z += sred[w].z; s.w += sred[w].w;
    }
    const float fn  = fmaxf(sqrtf(s.x), 1e-8f);
    const float cn  = fmaxf(sqrtf(s.y), 1e-8f);
    const float sim = s.z / (fn * cn);
    intra_partial[i] = s.w * expf(-sim);       // sq_err / exp(ALPHA*sim), ALPHA=1
    atomicAdd(&hist[lab], 1);
    s_inv = 1.0f / fn;
  }
  __syncthreads();
  const float inv = s_inv;
#pragma unroll
  for (int d = t; d < D_; d += 256)
    fhat[(size_t)i * D_ + d] = (_Float16)(f[d] * inv);
}

// ---------------------------------------------------------------------------
// Tiled symmetric GEMM + masked hinge reduction.
// 128 threads = 4 wave32 waves; wave (wm,wn) owns 64x64 = 4x4 WMMA tiles.
// Double-buffered LDS; async fills overlap the WMMA stream.
// ---------------------------------------------------------------------------
__global__ __launch_bounds__(128) void bdc_gemm(
    const _Float16* __restrict__ fhat, const int* __restrict__ labels,
    float* __restrict__ hinge_partial) {
  const int bj = blockIdx.x, bi = blockIdx.y;
  const int blin = bi * NTILE + bj;
  if (bi > bj) {
    if (threadIdx.x == 0) hinge_partial[blin] = 0.f;
    return;
  }

  __shared__ __align__(16) _Float16 As[2][BM * LDSK];
  __shared__ __align__(16) _Float16 Bs[2][BM * LDSK];
  __shared__ int   Li[BM], Lj[BM];
  __shared__ float wsum[4];

  const int t = threadIdx.x;
  const int wave = t >> 5, lane = t & 31;
  const int wm = wave & 1;        // 0..1 -> 64-row group
  const int wn = wave >> 1;       // 0..1 -> 64-col group
  const int i0 = bi * BM, j0 = bj * BM;

  Li[t] = labels[i0 + t];
  Lj[t] = labels[j0 + t];

  v8f acc[4][4] = {};

  // prologue fill of buffer 0
  stage_load(fhat + (size_t)i0 * D_, As[0], t);
  stage_load(fhat + (size_t)j0 * D_, Bs[0], t);

  for (int k0 = 0; k0 < D_; k0 += BK) {
    const int cur = (k0 / BK) & 1;
#ifdef ASYNC_OK
    asm volatile("s_wait_asynccnt 0" ::: "memory");
#endif
    __syncthreads();    // buf[cur] published; everyone done reading buf[cur^1]
    if (k0 + BK < D_) {
      stage_load(fhat + (size_t)i0 * D_ + (k0 + BK), As[cur ^ 1], t);
      stage_load(fhat + (size_t)j0 * D_ + (k0 + BK), Bs[cur ^ 1], t);
    }
#pragma unroll
    for (int kk = 0; kk < BK; kk += 32) {
      v16h a[4], b[4];
#pragma unroll
      for (int mt = 0; mt < 4; ++mt)
        a[mt] = load_frag(&As[cur][(wm * 64 + mt * 16) * LDSK + kk], lane);
#pragma unroll
      for (int nt = 0; nt < 4; ++nt)
        b[nt] = load_frag(&Bs[cur][(wn * 64 + nt * 16) * LDSK + kk], lane);
#pragma unroll
      for (int mt = 0; mt < 4; ++mt)
#pragma unroll
        for (int nt = 0; nt < 4; ++nt)
          acc[mt][nt] = __builtin_amdgcn_wmma_f32_16x16x32_f16(
              false, a[mt], false, b[nt], (short)0, acc[mt][nt], false, false);
    }
  }

  // Epilogue: C/D layout -> element e of v8f: M = e + 8*(lane>>4), N = lane&15
  const float wgt   = (bi == bj) ? 1.0f : 2.0f;
  const int   colN  = lane & 15;
  const int   mhalf = (lane >> 4) * 8;
  float hsum = 0.f;
#pragma unroll
  for (int mt = 0; mt < 4; ++mt) {
#pragma unroll
    for (int nt = 0; nt < 4; ++nt) {
      const int lj = Lj[wn * 64 + nt * 16 + colN];
#pragma unroll
      for (int e = 0; e < 8; ++e) {
        const int li = Li[wm * 64 + mt * 16 + mhalf + e];
        const float h = 0.5f - acc[mt][nt][e];      // MARGIN - sim
        hsum += (li != lj && h > 0.f) ? h : 0.f;
      }
    }
  }
  hsum = wave_sum(hsum) * wgt;
  if (lane == 0) wsum[wave] = hsum;
  __syncthreads();
  if (t == 0) {
    float s = 0.f;
#pragma unroll
    for (int w = 0; w < 4; ++w) s += wsum[w];
    hinge_partial[blin] = s;
  }
}

// ---------------------------------------------------------------------------
// Final combine: intra mean, hinge sum / n_pairs, n_pairs = B^2 - sum c^2.
// ---------------------------------------------------------------------------
__global__ __launch_bounds__(1024) void bdc_final(
    const float* __restrict__ intra_partial, const float* __restrict__ hinge_partial,
    const int* __restrict__ hist, float* __restrict__ out) {
  const int t = threadIdx.x;
  float isum = 0.f;
  for (int k = t; k < B_; k += 1024) isum += intra_partial[k];
  float hsum = 0.f;
  for (int k = t; k < NBLK; k += 1024) hsum += hinge_partial[k];
  double csq = 0.0;
  for (int k = t; k < C_; k += 1024) {
    const double h = (double)hist[k];
    csq += h * h;
  }
  isum = wave_sum(isum);
  hsum = wave_sum(hsum);
  csq  = wave_sum_d(csq);

  __shared__ float  sI[32], sH[32];
  __shared__ double sC[32];
  const int wave = t >> 5, lane = t & 31;
  if (lane == 0) { sI[wave] = isum; sH[wave] = hsum; sC[wave] = csq; }
  __syncthreads();
  if (t == 0) {
    float I = 0.f, H = 0.f; double Cq = 0.0;
#pragma unroll
    for (int w = 0; w < 32; ++w) { I += sI[w]; H += sH[w]; Cq += sC[w]; }
    double n_pairs = (double)B_ * (double)B_ - Cq;
    if (n_pairs < 1.0) n_pairs = 1.0;
    const float intra = I / (float)B_;
    const float adv   = (float)((double)H / n_pairs);
    out[0] = intra + 0.5f * adv;   // LAMBDA_ADV = 0.5
  }
}

// ---------------------------------------------------------------------------
extern "C" void kernel_launch(void* const* d_in, const int* in_sizes, int n_in,
                              void* d_out, int out_size, void* d_ws, size_t ws_size,
                              hipStream_t stream) {
  (void)in_sizes; (void)n_in; (void)out_size; (void)ws_size;
  const float* feat    = (const float*)d_in[0];
  const int*   labels  = (const int*)d_in[1];
  const float* centers = (const float*)d_in[2];

  char* ws = (char*)d_ws;
  _Float16* fhat = (_Float16*)ws;                                          // 16 MB
  float* intra_p = (float*)(ws + (size_t)B_ * D_ * sizeof(_Float16));      // 32 KB
  float* hinge_p = (float*)((char*)intra_p + (size_t)B_ * sizeof(float));  // 16 KB
  int*   hist    = (int*)((char*)hinge_p + (size_t)NBLK * sizeof(float));  // 4 KB

  bdc_zero_hist<<<1, 1024, 0, stream>>>(hist);
  bdc_row<<<B_, 256, 0, stream>>>(feat, labels, centers, fhat, intra_p, hist);
  dim3 grid(NTILE, NTILE);
  bdc_gemm<<<grid, 128, 0, stream>>>(fhat, labels, hinge_p);
  bdc_final<<<1, 1024, 0, stream>>>(intra_p, hinge_p, hist, (float*)d_out);
}
